// MiniPhiRNN_7980049236539
// MI455X (gfx1250) — matrix operand order repaired and checked by reading, not compiled, not run on hardware
//
#include <hip/hip_runtime.h>

typedef __attribute__((ext_vector_type(16))) __bf16 v16bf;
typedef __attribute__((ext_vector_type(8)))  __bf16 v8bf;
typedef __attribute__((ext_vector_type(8)))  float  v8f;

#define D      128
#define NB     8
#define DEPTH  3
#define TLEN   512
#define BTILE  16
#define SPITCH 132   // state row pitch (f32)
#define APITCH 136   // A-tile row pitch (bf16)
#define WPITCH 136   // weight row pitch (bf16)

// ---- dynamic LDS layout (bytes) ----
constexpr int OFF_WL  = 0;
constexpr int SZ_WL   = DEPTH * D * WPITCH * 2;      // 104448
constexpr int OFF_ST  = OFF_WL + SZ_WL;
constexpr int SZ_ST   = BTILE * SPITCH * 4;          // 8448
constexpr int OFF_AT  = OFF_ST + SZ_ST;
constexpr int SZ_AT   = BTILE * APITCH * 2;          // 4352
constexpr int OFF_XB  = OFF_AT + SZ_AT;
constexpr int SZ_XB   = BTILE * NB * 4;              // 512
constexpr int OFF_WI  = OFF_XB + SZ_XB;
constexpr int SZ_WI   = D * NB * 4;                  // 4096
constexpr int OFF_WO  = OFF_WI + SZ_WI;
constexpr int SZ_WO   = NB * D * 4;                  // 4096
constexpr int OFF_BI  = OFF_WO + SZ_WO;
constexpr int OFF_GSN = OFF_BI  + D * 4;
constexpr int OFF_BSN = OFF_GSN + D * 4;
constexpr int OFF_BL  = OFF_BSN + D * 4;
constexpr int OFF_GLN = OFF_BL  + DEPTH * D * 4;
constexpr int OFF_BLN = OFF_GLN + DEPTH * D * 4;
constexpr int OFF_BO  = OFF_BLN + DEPTH * D * 4;
constexpr int SMEM_BYTES = OFF_BO + NB * 4;          // ~132 KB

__device__ __forceinline__ unsigned short f2bf(float f) {
    unsigned int u = __float_as_uint(f);
    u += 0x7FFFu + ((u >> 16) & 1u);                 // round-to-nearest-even
    return (unsigned short)(u >> 16);
}

// Branchless folded-cosine activation: compute everything, select at the end.
__device__ __forceinline__ float c19_act(float xv) {
    const float INVPI  = 0.3183098861837907f;
    const float PIF    = 3.14159265358979f;
    const float LBOUND = 18.84955592153876f;         // 6*pi
    float scaled = xv * INVPI;
    float n   = floorf(scaled);
    float tt  = scaled - n;
    float h   = tt * (1.0f - tt);
    float r   = n - 2.0f * floorf(n * 0.5f);         // remainder(n,2) in [0,2)
    float sh  = (r < 1.0f) ? h : -h;                 // v_cndmask
    float core = PIF * fmaf(4.0f * h, h, sh);
    float hi  = xv - LBOUND;
    float lo  = xv + LBOUND;
    float res = (xv >= LBOUND) ? hi : core;          // v_cndmask
    res       = (xv <= -LBOUND) ? lo : res;          // v_cndmask
    return res;
}

extern "C" __global__ __launch_bounds__(256)
void miniphi_rnn_kernel(const float* __restrict__ x,
                        const float* __restrict__ Wi,
                        const float* __restrict__ bi,
                        const float* __restrict__ Wo,
                        const float* __restrict__ bo,
                        const float* __restrict__ g_sn,
                        const float* __restrict__ b_sn,
                        const float* __restrict__ Wl,
                        const float* __restrict__ bl,
                        const float* __restrict__ g_ln,
                        const float* __restrict__ b_ln,
                        float* __restrict__ out) {
    extern __shared__ char smem[];
    unsigned short* wlLds = (unsigned short*)(smem + OFF_WL);
    float*          stLds = (float*)(smem + OFF_ST);
    unsigned short* aLds  = (unsigned short*)(smem + OFF_AT);
    float*          xbLds = (float*)(smem + OFF_XB);
    float*          wiLds = (float*)(smem + OFF_WI);
    float*          woLds = (float*)(smem + OFF_WO);
    float*          biLds = (float*)(smem + OFF_BI);
    float*          gsnL  = (float*)(smem + OFF_GSN);
    float*          bsnL  = (float*)(smem + OFF_BSN);
    float*          blLds = (float*)(smem + OFF_BL);
    float*          glnL  = (float*)(smem + OFF_GLN);
    float*          blnL  = (float*)(smem + OFF_BLN);
    float*          boLds = (float*)(smem + OFF_BO);

    const int tid  = threadIdx.x;
    const int wave = tid >> 5;
    const int lane = tid & 31;
    const int b0   = blockIdx.x * BTILE;

    // ---- one-time: stage weights in LDS (Wl as bf16) ----
    for (int idx = tid; idx < DEPTH * D * D; idx += 256) {
        int l = idx / (D * D);
        int rem = idx - l * D * D;
        int n = rem >> 7, k = rem & (D - 1);
        wlLds[(l * D + n) * WPITCH + k] = f2bf(Wl[idx]);
    }
    for (int idx = tid; idx < D * NB; idx += 256) { wiLds[idx] = Wi[idx]; woLds[idx] = Wo[idx]; }
    for (int idx = tid; idx < D; idx += 256) { biLds[idx] = bi[idx]; gsnL[idx] = g_sn[idx]; bsnL[idx] = b_sn[idx]; }
    for (int idx = tid; idx < DEPTH * D; idx += 256) { blLds[idx] = bl[idx]; glnL[idx] = g_ln[idx]; blnL[idx] = b_ln[idx]; }
    if (tid < NB) boLds[tid] = bo[tid];
    for (int idx = tid; idx < BTILE * SPITCH; idx += 256) stLds[idx] = 0.f;   // h0 = 0
    __syncthreads();

    // half-wave -> row mapping for layernorm work
    const int row   = 2 * wave + (lane >> 4);
    const int li    = lane & 15;
    const int kbase = li * 8;

    // ---- software-pipelined x_t: hold next timestep's slice in a register ----
    const int xm = tid >> 3, xj = tid & 7;           // valid when tid < 128
    float xreg = 0.f;
    if (tid < BTILE * NB)
        xreg = x[((size_t)(b0 + xm) * TLEN + 0) * NB + xj];

    for (int t = 0; t < TLEN; ++t) {
        if (tid < BTILE * NB) xbLds[tid] = xreg;     // commit x_t
        __syncthreads();
        // issue x_{t+1} load now: latency overlaps the whole timestep body
        if (t + 1 < TLEN && tid < BTILE * NB)
            xreg = x[((size_t)(b0 + xm) * TLEN + (t + 1)) * NB + xj];

        // ---- state = 0.618*LN(h)*g_sn+b_sn + 0.382*(x_t @ Wi^T + bi) ----
        {
            float* sr = stLds + row * SPITCH;
            float v[8], s1 = 0.f, s2 = 0.f;
            #pragma unroll
            for (int e = 0; e < 8; ++e) { v[e] = sr[kbase + e]; s1 += v[e]; s2 += v[e] * v[e]; }
            #pragma unroll
            for (int m = 8; m >= 1; m >>= 1) { s1 += __shfl_xor(s1, m, 32); s2 += __shfl_xor(s2, m, 32); }
            float mean = s1 * (1.f / D);
            float var  = fmaxf(s2 * (1.f / D) - mean * mean, 0.f);
            float inv  = rsqrtf(var + 1e-5f);
            const float* xr = xbLds + row * NB;
            float xv[NB];
            #pragma unroll
            for (int j = 0; j < NB; ++j) xv[j] = xr[j];
            #pragma unroll
            for (int e = 0; e < 8; ++e) {
                int k = kbase + e;
                float ln  = (v[e] - mean) * inv * gsnL[k] + bsnL[k];
                const float* wr = wiLds + k * NB;
                float inp = biLds[k];
                #pragma unroll
                for (int j = 0; j < NB; ++j) inp += xv[j] * wr[j];
                sr[k] = 0.618f * ln + 0.382f * inp;
            }
        }
        __syncthreads();

        // ---- 3 residual layers: state += act(LN(state) @ Wl^T + bl) ----
        #pragma unroll
        for (int l = 0; l < DEPTH; ++l) {
            {   // LN -> bf16 A tile
                float* sr = stLds + row * SPITCH;
                float v[8], s1 = 0.f, s2 = 0.f;
                #pragma unroll
                for (int e = 0; e < 8; ++e) { v[e] = sr[kbase + e]; s1 += v[e]; s2 += v[e] * v[e]; }
                #pragma unroll
                for (int m = 8; m >= 1; m >>= 1) { s1 += __shfl_xor(s1, m, 32); s2 += __shfl_xor(s2, m, 32); }
                float mean = s1 * (1.f / D);
                float var  = fmaxf(s2 * (1.f / D) - mean * mean, 0.f);
                float inv  = rsqrtf(var + 1e-5f);
                #pragma unroll
                for (int e = 0; e < 8; ++e) {
                    int k = kbase + e;
                    aLds[row * APITCH + k] = f2bf((v[e] - mean) * inv * glnL[l * D + k] + blnL[l * D + k]);
                }
            }
            __syncthreads();

            // wave w computes output columns [16w, 16w+16): 4 chained WMMAs over K=128
            v8f acc = {};
            const int arow = lane & 15;
            const int hi   = lane >> 4;
            #pragma unroll
            for (int ks = 0; ks < 4; ++ks) {
                // A frag (16-bit 16x32 layout): lane holds row arow, K chunks {32ks+8hi, 32ks+16+8hi}
                const unsigned short* ap = aLds + arow * APITCH + ks * 32 + hi * 8;
                v8bf a0 = *(const v8bf*)(const void*)(ap);
                v8bf a1 = *(const v8bf*)(const void*)(ap + 16);
                // B frag: lane holds column (16w+arow), K chunks {32ks+8hi, 32ks+16+8hi} in Wl row
                const unsigned short* bp = wlLds + (l * D + (wave * 16 + arow)) * WPITCH + ks * 32 + hi * 8;
                v8bf q0 = *(const v8bf*)(const void*)(bp);
                v8bf q1 = *(const v8bf*)(const void*)(bp + 16);
                v16bf afrag, bfrag;
                #pragma unroll
                for (int e = 0; e < 8; ++e) {
                    afrag[e] = a0[e]; afrag[8 + e] = a1[e];
                    bfrag[e] = q0[e]; bfrag[8 + e] = q1[e];
                }
                acc = __builtin_amdgcn_wmma_f32_16x16x32_bf16(
                          false, afrag, false, bfrag, (short)0, acc, false, false);
            }
            // D layout: lane<16 -> (M=r, N=lane); lane>=16 -> (M=8+r, N=lane-16)
            const int col  = wave * 16 + (lane & 15);
            const int mb   = (lane >> 4) * 8;
            const float bias = blLds[l * D + col];
            float oldv[8];
            #pragma unroll
            for (int r = 0; r < 8; ++r) oldv[r] = stLds[(mb + r) * SPITCH + col];
            #pragma unroll
            for (int r = 0; r < 8; ++r) oldv[r] += c19_act(acc[r] + bias);
            #pragma unroll
            for (int r = 0; r < 8; ++r) stLds[(mb + r) * SPITCH + col] = oldv[r];
            __syncthreads();
        }

        // ---- out = state @ Wo^T + bo : 128 outputs, 2 lanes per dot ----
        {
            int p    = tid >> 1;                     // output index 0..127
            int half = tid & 1;                      // k-range half
            int m    = p >> 3, j = p & 7;
            const float* sr = stLds + m * SPITCH + half * 64;
            const float* wr = woLds + j * D + half * 64;
            float o = 0.f;
            #pragma unroll 16
            for (int k = 0; k < 64; ++k) o += sr[k] * wr[k];
            o += __shfl_xor(o, 1, 32);               // combine lane pair
            if (half == 0)
                out[((size_t)(b0 + m) * TLEN + t) * NB + j] = o + boLds[j];
        }
        __syncthreads();
    }
}

extern "C" void kernel_launch(void* const* d_in, const int* in_sizes, int n_in,
                              void* d_out, int out_size, void* d_ws, size_t ws_size,
                              hipStream_t stream) {
    const float* x   = (const float*)d_in[0];
    const float* Wi  = (const float*)d_in[1];
    const float* bi  = (const float*)d_in[2];
    const float* Wo  = (const float*)d_in[3];
    const float* bo  = (const float*)d_in[4];
    const float* gsn = (const float*)d_in[5];
    const float* bsn = (const float*)d_in[6];
    const float* Wl  = (const float*)d_in[7];
    const float* bl  = (const float*)d_in[8];
    const float* gln = (const float*)d_in[9];
    const float* bln = (const float*)d_in[10];
    float* out = (float*)d_out;

    const int Btotal = in_sizes[0] / (TLEN * NB);   // 1024
    const int grid   = Btotal / BTILE;              // 64 persistent workgroups

    hipLaunchKernelGGL(miniphi_rnn_kernel, dim3(grid), dim3(256), SMEM_BYTES, stream,
                       x, Wi, bi, Wo, bo, gsn, bsn, Wl, bl, gln, bln, out);
}
